// ResidualGatedBlock_26706106646671
// MI455X (gfx1250) — compile-verified, matrix operand order checked by reading, not run on hardware
//
#include <hip/hip_runtime.h>
#include <hip/hip_bf16.h>

typedef __bf16 bf16_t;
typedef __attribute__((ext_vector_type(16))) __bf16 v16bf;
typedef __attribute__((ext_vector_type(8)))  __bf16 v8bf;
typedef __attribute__((ext_vector_type(8)))  float  v8f;

#define RC   64
#define SC   128
#define DC   64
#define KW   2
#define DIL  8
#define BB   32
#define TT   16384
#define NT   128                 // time columns per workgroup
#define XROWS (NT + DIL)         // 136 staged time rows (incl. left halo)
#define XPITCH 80                // halves per time row (160B, 16B aligned)
#define CPITCH 72                // halves per concat row (144B, 16B aligned)
#define WS_P2_OFF 16384          // halves: phase-1 frags = 32 frags * 512 halves

// Map (row-in-tile, k-in-32-step) -> (lane, half) for 16-bit WMMA A/B fragments.
// Layout per ISA 7.12.2: lanes 0-15 hold K = 8g+{0..7} (halves 0-7) and
// K = 16+8g+{0..7} (halves 8-15), g = lane/16.
__device__ __host__ inline void frag_slot(int mloc, int kloc, int& lane, int& h) {
    int g, hh;
    if (kloc < 16) { g = kloc >> 3;            hh = kloc & 7; }
    else           { int k2 = kloc - 16; g = k2 >> 3; hh = 8 + (k2 & 7); }
    lane = mloc + 16 * g;
    h = hh;
}

// ---------------------------------------------------------------------------
// Prep kernel: pack all weights (f32 -> bf16) into WMMA A-fragment order.
// Region 1 (phase 1): 128 rows (gate 0..63, filter 64..127) x K=128
//   K = 0..63  -> tap k=1 (current t), channel K
//   K = 64..127-> tap k=0 (t - DIL),   channel K-64
// Region 2 (phase 2): 192 rows (skip 0..127, out 128..191) x K=64 (concat ch)
// Fragment f stored at f*512 halves; lane l owns halves [l*16, l*16+16).
// ---------------------------------------------------------------------------
__global__ void prep_weights(const float* __restrict__ wg,
                             const float* __restrict__ wf,
                             const float* __restrict__ wsk,
                             const float* __restrict__ wo,
                             bf16_t* __restrict__ ws) {
    int idx = blockIdx.x * blockDim.x + threadIdx.x;
    if (idx < 128 * 128) {
        int m  = idx >> 7;
        int kk = idx & 127;
        int tap = (kk < 64) ? 1 : 0;
        int ci  = kk & 63;
        float v = (m < 64) ? wg[m * (RC * KW) + ci * KW + tap]
                           : wf[(m - 64) * (RC * KW) + ci * KW + tap];
        int mt = m >> 4, ks = kk >> 5;
        int lane, h; frag_slot(m & 15, kk & 31, lane, h);
        ws[(mt * 4 + ks) * 512 + lane * 16 + h] = (bf16_t)v;
    }
    int idx2 = idx - 128 * 128;
    if (idx2 >= 0 && idx2 < 192 * 64) {
        int m  = idx2 >> 6;
        int kk = idx2 & 63;
        float v = (m < 128) ? wsk[m * DC + kk] : wo[(m - 128) * DC + kk];
        int mt = m >> 4, ks = kk >> 5;
        int lane, h; frag_slot(m & 15, kk & 31, lane, h);
        ws[WS_P2_OFF + (mt * 2 + ks) * 512 + lane * 16 + h] = (bf16_t)v;
    }
}

// ---------------------------------------------------------------------------
// Main fused kernel: one block = (batch b, 128-time tile). 8 waves; wave w owns
// columns [16w, 16w+16) of the tile for both GEMM phases.
// ---------------------------------------------------------------------------
__global__ __launch_bounds__(256, 1)
void wavenet_block(const float* __restrict__ x,
                   const float* __restrict__ b_gate,
                   const float* __restrict__ b_filter,
                   const float* __restrict__ b_skip,
                   const float* __restrict__ b_out,
                   const bf16_t* __restrict__ wfrag,
                   float* __restrict__ out_res,
                   float* __restrict__ out_skip) {
    __shared__ __align__(16) bf16_t sX[XROWS * XPITCH];  // x tile, time-major
    __shared__ __align__(16) bf16_t sC[NT * CPITCH];     // concat, time-major

    const int t0  = blockIdx.x * NT;
    const int b   = blockIdx.y;
    const int tid = threadIdx.x;

    // ---- stage x[b, :, t0-8 .. t0+NT) into LDS as bf16 (time-major) ----
    for (int idx = tid; idx < 64 * 144; idx += 256) {
        int c  = idx / 144;
        int tl = idx - c * 144;
        if (tl < XROWS) {
            int t = t0 - DIL + tl;
            float v = (t >= 0) ? x[((size_t)b * RC + c) * TT + t] : 0.0f;
            sX[tl * XPITCH + c] = (bf16_t)v;
        }
    }
    __syncthreads();

    const int wave  = tid >> 5;
    const int lane  = tid & 31;
    const int nloc  = lane & 15;          // column within 16-col slice
    const int g     = lane >> 4;          // lane half
    const int ncol  = wave * 16 + nloc;   // column within NT tile
    const int tglob = t0 + ncol;

    // ---- phase 1 B fragments: 4 k-steps of 32 over stacked K=128 ----
    v16bf Bf[4];
#pragma unroll
    for (int ks = 0; ks < 4; ++ks) {
        int k0 = ks * 32 + 8 * g;          // first run of 8 K-values
        int k1 = ks * 32 + 16 + 8 * g;     // second run of 8 K-values
        int tl0 = (k0 < 64) ? (ncol + DIL) : ncol;   // K<64 -> x[t], else x[t-8]
        int tl1 = (k1 < 64) ? (ncol + DIL) : ncol;
        v8bf r0 = *(const v8bf*)&sX[tl0 * XPITCH + (k0 & 63)];
        v8bf r1 = *(const v8bf*)&sX[tl1 * XPITCH + (k1 & 63)];
        Bf[ks] = __builtin_shufflevector(r0, r1, 0, 1, 2, 3, 4, 5, 6, 7,
                                                 8, 9, 10, 11, 12, 13, 14, 15);
    }

    // ---- phase 1 GEMM: mt 0..3 = gate rows, mt 4..7 = filter rows ----
    v8f acc[8];
#pragma unroll
    for (int mt = 0; mt < 8; ++mt) {
        v8f a = {};
#pragma unroll
        for (int ks = 0; ks < 4; ++ks) {
            const bf16_t* ap = wfrag + (mt * 4 + ks) * 512 + lane * 16;
            v8bf a0 = *(const v8bf*)ap;
            v8bf a1 = *(const v8bf*)(ap + 8);
            v16bf A = __builtin_shufflevector(a0, a1, 0, 1, 2, 3, 4, 5, 6, 7,
                                                      8, 9, 10, 11, 12, 13, 14, 15);
            a = __builtin_amdgcn_wmma_f32_16x16x32_bf16(
                    false, A, false, Bf[ks], (short)0, a, false, false);
        }
        acc[mt] = a;
    }

    // ---- fuse: sigmoid(gate) * tanh(filter) -> concat (bf16, LDS) ----
#pragma unroll
    for (int p = 0; p < 4; ++p) {
#pragma unroll
        for (int r = 0; r < 8; ++r) {
            int m = 16 * p + 8 * g + r;   // D-layout: row = r + 8*(lane/16)
            float gv = acc[p][r] + b_gate[m];
            float fv = acc[4 + p][r] + b_filter[m];
            gv = 1.0f / (1.0f + __expf(-gv));
            fv = tanhf(fv);
            sC[ncol * CPITCH + m] = (bf16_t)(gv * fv);
        }
    }
    // Each wave reads back only its own 16-column slice; per-wave DS ops are
    // in-order, so no workgroup barrier is required here.

    // ---- phase 2 B fragments: K=64 concat channels, 2 k-steps ----
    v16bf Cf[2];
#pragma unroll
    for (int ks = 0; ks < 2; ++ks) {
        v8bf r0 = *(const v8bf*)&sC[ncol * CPITCH + (ks * 32 + 8 * g)];
        v8bf r1 = *(const v8bf*)&sC[ncol * CPITCH + (ks * 32 + 16 + 8 * g)];
        Cf[ks] = __builtin_shufflevector(r0, r1, 0, 1, 2, 3, 4, 5, 6, 7,
                                                 8, 9, 10, 11, 12, 13, 14, 15);
    }

    // ---- phase 2 GEMM: mt 0..7 = skip rows, mt 8..11 = residual-out rows ----
    const bf16_t* w2 = wfrag + WS_P2_OFF;
#pragma unroll
    for (int mt = 0; mt < 12; ++mt) {
        v8f a = {};
#pragma unroll
        for (int ks = 0; ks < 2; ++ks) {
            const bf16_t* ap = w2 + (mt * 2 + ks) * 512 + lane * 16;
            v8bf a0 = *(const v8bf*)ap;
            v8bf a1 = *(const v8bf*)(ap + 8);
            v16bf A = __builtin_shufflevector(a0, a1, 0, 1, 2, 3, 4, 5, 6, 7,
                                                      8, 9, 10, 11, 12, 13, 14, 15);
            a = __builtin_amdgcn_wmma_f32_16x16x32_bf16(
                    false, A, false, Cf[ks], (short)0, a, false, false);
        }
        if (mt < 8) {
#pragma unroll
            for (int r = 0; r < 8; ++r) {
                int m = 16 * mt + 8 * g + r;
                out_skip[((size_t)b * SC + m) * TT + tglob] = a[r] + b_skip[m];
            }
        } else {
            int mo = mt - 8;
#pragma unroll
            for (int r = 0; r < 8; ++r) {
                int m = 16 * mo + 8 * g + r;
                size_t xi = ((size_t)b * RC + m) * TT + tglob;
                out_res[xi] = a[r] + b_out[m] + x[xi];   // residual add
            }
        }
    }
}

extern "C" void kernel_launch(void* const* d_in, const int* in_sizes, int n_in,
                              void* d_out, int out_size, void* d_ws, size_t ws_size,
                              hipStream_t stream) {
    (void)in_sizes; (void)n_in; (void)out_size; (void)ws_size;
    const float* x        = (const float*)d_in[0];
    const float* w_gate   = (const float*)d_in[1];
    const float* b_gate   = (const float*)d_in[2];
    const float* w_filter = (const float*)d_in[3];
    const float* b_filter = (const float*)d_in[4];
    const float* w_skip   = (const float*)d_in[5];
    const float* b_skip   = (const float*)d_in[6];
    const float* w_out    = (const float*)d_in[7];
    const float* b_out    = (const float*)d_in[8];

    float* out_res  = (float*)d_out;                       // outputs [B,RC,T]
    float* out_skip = out_res + (size_t)BB * RC * TT;      // skip    [B,SC,T]
    bf16_t* wfrag   = (bf16_t*)d_ws;                       // ~57 KB used

    int prep_n = 128 * 128 + 192 * 64;
    prep_weights<<<(prep_n + 255) / 256, 256, 0, stream>>>(w_gate, w_filter,
                                                           w_skip, w_out, wfrag);
    dim3 grid(TT / NT, BB);
    wavenet_block<<<grid, 256, 0, stream>>>(x, b_gate, b_filter, b_skip, b_out,
                                            wfrag, out_res, out_skip);
}
